// TeacherGruActorCritic_80796924772448
// MI455X (gfx1250) — matrix operand order repaired and checked by reading, not compiled, not run on hardware
//
#include <hip/hip_runtime.h>

typedef __attribute__((ext_vector_type(16))) _Float16 v16h;
typedef __attribute__((ext_vector_type(8)))  _Float16 v8h;
typedef __attribute__((ext_vector_type(8)))  float    v8f;
typedef __attribute__((ext_vector_type(4)))  unsigned int u32x4;
typedef __attribute__((ext_vector_type(8)))  int          i32x8;
typedef __attribute__((ext_vector_type(4)))  int          i32x4;

#define DEV static __device__ __forceinline__

static constexpr int Bn = 4096, Tn = 64, Fn = 256, Hn = 256, An = 12;

// ---------------- WMMA fragment helpers (CDNA5 wave32 layouts, ISA 7.12.2) ----

// A fragment 16x32 f16 from row-major f16 matrix, row stride ld (halfs).
// lane l<16: M=m0+l, K = k0..k0+7 (elts 0..7) and k0+16..k0+23 (elts 8..15)
// lane l>=16: M=m0+l-16, K = k0+8..15 and k0+24..31
DEV v16h frag_a_f16(const _Float16* A, int m0, int k0, int ld) {
  int l = threadIdx.x & 31;
  const _Float16* p = A + (size_t)(m0 + (l & 15)) * ld + k0 + ((l & 16) ? 8 : 0);
  v8h lo = *(const v8h*)(p);
  v8h hi = *(const v8h*)(p + 16);
  v16h f;
#pragma unroll
  for (int i = 0; i < 8; ++i) { f[i] = lo[i]; f[i + 8] = hi[i]; }
  return f;
}

// Same A fragment but sourced from row-major f32 (converted on the fly).
DEV v16h frag_a_f32(const float* A, int m0, int k0, int ld) {
  int l = threadIdx.x & 31;
  const float* p = A + (size_t)(m0 + (l & 15)) * ld + k0 + ((l & 16) ? 8 : 0);
  v8f lo = *(const v8f*)(p);
  v8f hi = *(const v8f*)(p + 16);
  v16h f;
#pragma unroll
  for (int i = 0; i < 8; ++i) { f[i] = (_Float16)lo[i]; f[i + 8] = (_Float16)hi[i]; }
  return f;
}

// B fragment 32x16 where B = W^T, W row-major [N,K], row stride ld (halfs).
// lane l<16: col N=n0+l, K=k0..k0+7 and k0+16..23 ; lane l>=16: col N=n0+l-16, K=k0+8..15 and k0+24..31
DEV v16h frag_b_w(const _Float16* W, int n0, int k0, int ld) {
  int l = threadIdx.x & 31;
  const _Float16* p = W + (size_t)(n0 + (l & 15)) * ld + k0 + ((l & 16) ? 8 : 0);
  v8h lo = *(const v8h*)(p);
  v8h hi = *(const v8h*)(p + 16);
  v16h f;
#pragma unroll
  for (int i = 0; i < 8; ++i) { f[i] = lo[i]; f[i + 8] = hi[i]; }
  return f;
}

DEV v8f wmma16(v16h a, v16h b, v8f c) {
  return __builtin_amdgcn_wmma_f32_16x16x32_f16(false, a, false, b, (short)0, c,
                                                false, false);
}

DEV float sigmoidf_(float x) { return 1.0f / (1.0f + __expf(-x)); }

// ---------------- TDM: async 2D tile load (global f16 -> LDS) ----------------
// Loads a [rows=32 x cols=768] f16 tile, row stride rstride (elements), into LDS.
// D# layout per CDNA5 ISA 8.3/8.4 (group0 128b, group1 256b; groups 2/3 unused).
// clang-23 toolchain: 6-arg builtin (u32x4, i32x8, i32x4, i32x4, i32x8, i32 cpol).
DEV void tdm_load_xg_tile(const _Float16* gsrc, void* lds_dst, int rstride) {
  unsigned long long ga = (unsigned long long)(uintptr_t)gsrc;
  unsigned lds = (unsigned)(uintptr_t)lds_dst;  // flat LDS addr truncates to offset
  u32x4 g0;
  g0[0] = 1u;                                            // count=1 (valid), user mode
  g0[1] = lds;                                           // lds_addr
  g0[2] = (unsigned)ga;                                  // global_addr[31:0]
  g0[3] = (unsigned)((ga >> 32) & 0x01FFFFFFull)         // global_addr[56:32]
        | (2u << 30);                                    // type=2 ("image")
  i32x8 g1;
  g1[0] = (1 << 16);        // workgroup_mask=0, data_size=1 (2 bytes)
  g1[1] = (768 << 16);      // tensor_dim0 = 768     (bits 79:48)
  g1[2] = (32 << 16);       // tensor_dim1 = 32      (bits 111:80)
  g1[3] = (768 << 16);      // tile_dim0   = 768     (bits 127:112)
  g1[4] = 32;               // tile_dim1   = 32, tile_dim2 = 0
  g1[5] = rstride;          // tensor_dim0_stride[31:0]  (bits 207:160)
  g1[6] = 0;                // tensor_dim0_stride[47:32], tensor_dim1_stride lo
  g1[7] = 0;
  i32x4 z4 = {0, 0, 0, 0};
  i32x8 z8 = {0, 0, 0, 0, 0, 0, 0, 0};
  __builtin_amdgcn_tensor_load_to_lds(g0, g1, z4, z4, z8, 0);
}

// ---------------- weight conversion + std ------------------------------------

__global__ void k_convert_pad(const float* __restrict__ src, _Float16* __restrict__ dst,
                              int rows_src, int cols, int total) {
  int i = blockIdx.x * blockDim.x + threadIdx.x;
  if (i >= total) return;
  dst[i] = ((i / cols) < rows_src) ? (_Float16)src[i] : (_Float16)0.0f;
}

__global__ void k_std(const float* __restrict__ log_std, float* __restrict__ outp) {
  int i = threadIdx.x;
  if (i < An) {
    float v = log_std[i];
    v = fminf(fmaxf(v, -2.0f), -0.5f);
    outp[i] = __expf(v);
  }
}

// ---------------- kernel 1: xg = feat @ W_ih^T + b_ih  (f16 out) -------------
__global__ __launch_bounds__(256) void k_xg(const float* __restrict__ feat,
                                            const _Float16* __restrict__ Wih16,
                                            const float* __restrict__ b_ih,
                                            _Float16* __restrict__ xg) {
  const int m0 = blockIdx.x * 128 + (threadIdx.x >> 5) * 16;
  const int n0 = blockIdx.y * 128;
  v8f acc[8] = {};
#pragma unroll
  for (int k0 = 0; k0 < Fn; k0 += 32) {
    v16h a = frag_a_f32(feat, m0, k0, Fn);
#pragma unroll
    for (int j = 0; j < 8; ++j)
      acc[j] = wmma16(a, frag_b_w(Wih16, n0 + j * 16, k0, Fn), acc[j]);
  }
  const int l = threadIdx.x & 31;
  const int nn = l & 15, mo = (l & 16) ? 8 : 0;
#pragma unroll
  for (int j = 0; j < 8; ++j) {
    int n = n0 + j * 16 + nn;
    float bias = b_ih[n];
#pragma unroll
    for (int e = 0; e < 8; ++e) {
      size_t m = (size_t)(m0 + e + mo);
      xg[m * (3 * Hn) + n] = (_Float16)(acc[j][e] + bias);
    }
  }
}

// ---------------- kernel 2: GRU scan over T ----------------------------------
// 128 blocks x 256 threads. Block owns 32 batch rows; h lives in LDS; the
// per-step xg tile [32 x 768] f16 is TDM-prefetched into a double buffer.
__global__ __launch_bounds__(256) void k_scan(const _Float16* __restrict__ xg,
                                              const _Float16* __restrict__ Whh16,
                                              const float* __restrict__ b_hh,
                                              const float* __restrict__ hidden,
                                              const int* __restrict__ starts,
                                              _Float16* __restrict__ latent,
                                              float* __restrict__ h_last) {
  __shared__ float    h32[32][Hn];               // 32 KB master state
  __shared__ _Float16 h16[32][Hn];               // 16 KB masked WMMA operand
  __shared__ _Float16 xls[2][32][3 * Hn];        // 96 KB xg tile double buffer
  const int b0  = blockIdx.x * 32;
  const int tid = threadIdx.x;
  const int w   = tid >> 5, l = tid & 31;
  const int m0  = (w >> 2) * 16;                 // wave row tile: 0 or 16
  const int nI  = (w & 3) * 16;                  // wave col tile within 64-chunk
  const int nn  = l & 15, mo = (l & 16) ? 8 : 0;
  const int xstride = Tn * 3 * Hn;               // elements between batch rows

  if (tid < 32)  // wave 0 kicks off the tile for t=0 (TDM ignores EXEC)
    tdm_load_xg_tile(xg + (size_t)b0 * xstride, &xls[0][0][0], xstride);

  for (int i = tid; i < 32 * Hn; i += blockDim.x)
    h32[i >> 8][i & 255] = hidden[(size_t)(b0 + (i >> 8)) * Hn + (i & 255)];
  __syncthreads();

  for (int t = 0; t < Tn; ++t) {
    // start-masked f16 copy of h (GEMM operand for this step)
    for (int i = tid; i < 32 * Hn; i += blockDim.x) {
      int m = i >> 8, n = i & 255;
      int s = starts[(size_t)(b0 + m) * Tn + t];
      h16[m][n] = s ? (_Float16)0.0f : (_Float16)h32[m][n];
    }
    if (tid < 32) {  // prefetch t+1, then drain so tile t is resident
      if (t + 1 < Tn) {
        tdm_load_xg_tile(xg + (size_t)b0 * xstride + (size_t)(t + 1) * 3 * Hn,
                         &xls[(t + 1) & 1][0][0], xstride);
        __builtin_amdgcn_s_wait_tensorcnt(1);
      } else {
        __builtin_amdgcn_s_wait_tensorcnt(0);
      }
    }
    __syncthreads();
    const _Float16* xt = &xls[t & 1][0][0];

    // hoist the step's A fragments (reused by all 4 chunks x 3 gates)
    v16h afr[8];
#pragma unroll
    for (int kk = 0; kk < 8; ++kk)
      afr[kk] = frag_a_f16(&h16[0][0], m0, kk * 32, Hn);

#pragma unroll 1
    for (int c0 = 0; c0 < Hn; c0 += 64) {
      const int nb = c0 + nI;                    // this wave's output cols
      v8f ar = {}, az = {}, an = {};
#pragma unroll
      for (int kk = 0; kk < 8; ++kk) {
        ar = wmma16(afr[kk], frag_b_w(Whh16, 0 * Hn + nb, kk * 32, Hn), ar);
        az = wmma16(afr[kk], frag_b_w(Whh16, 1 * Hn + nb, kk * 32, Hn), az);
        an = wmma16(afr[kk], frag_b_w(Whh16, 2 * Hn + nb, kk * 32, Hn), an);
      }
      const int n = nb + nn;
      const float bhr = b_hh[n], bhz = b_hh[Hn + n], bhn = b_hh[2 * Hn + n];
#pragma unroll
      for (int e = 0; e < 8; ++e) {
        int m = m0 + e + mo;
        const _Float16* xrow = xt + (size_t)m * (3 * Hn);
        float xr = (float)xrow[n];
        float xz = (float)xrow[Hn + n];
        float xn = (float)xrow[2 * Hn + n];
        float r  = sigmoidf_(xr + ar[e] + bhr);
        float z  = sigmoidf_(xz + az[e] + bhz);
        float ng = tanhf(xn + r * (an[e] + bhn));
        int   s  = starts[(size_t)(b0 + m) * Tn + t];
        float ho = s ? 0.0f : h32[m][n];
        float hv = (1.0f - z) * ng + z * ho;
        h32[m][n] = hv;
        latent[((size_t)(b0 + m) * Tn + t) * Hn + n] = (_Float16)hv;
      }
    }
    __syncthreads();
  }
  for (int i = tid; i < 32 * Hn; i += blockDim.x)
    h_last[(size_t)(b0 + (i >> 8)) * Hn + (i & 255)] = h32[i >> 8][i & 255];
}

// ---------------- kernel 3: actor/critic MLP heads ---------------------------

DEV void layer256(const _Float16* Asrc, int arow0, int ald,
                  const _Float16* W, const float* bias,
                  _Float16* dst, int drow0) {
  const int l = threadIdx.x & 31;
  const int nn = l & 15, mo = (l & 16) ? 8 : 0;
  v16h a[8];
#pragma unroll
  for (int kk = 0; kk < 8; ++kk)
    a[kk] = frag_a_f16(Asrc, arow0, kk * 32, ald);
#pragma unroll 1
  for (int nh = 0; nh < Hn; nh += 128) {
    v8f acc[8] = {};
#pragma unroll
    for (int kk = 0; kk < 8; ++kk) {
#pragma unroll
      for (int j = 0; j < 8; ++j)
        acc[j] = wmma16(a[kk], frag_b_w(W, nh + j * 16, kk * 32, Hn), acc[j]);
    }
#pragma unroll
    for (int j = 0; j < 8; ++j) {
      int n = nh + j * 16 + nn;
      float bv = bias[n];
#pragma unroll
      for (int e = 0; e < 8; ++e)
        dst[(size_t)(drow0 + e + mo) * Hn + n] = (_Float16)tanhf(acc[j][e] + bv);
    }
  }
}

__global__ __launch_bounds__(256) void k_mlp(const _Float16* __restrict__ latent,
                                             const _Float16* __restrict__ Aw1,
                                             const float* __restrict__ Ab1,
                                             const _Float16* __restrict__ Aw2,
                                             const float* __restrict__ Ab2,
                                             const _Float16* __restrict__ Aw3p,
                                             const float* __restrict__ Ab3,
                                             const _Float16* __restrict__ Cw1,
                                             const float* __restrict__ Cb1,
                                             const _Float16* __restrict__ Cw2,
                                             const float* __restrict__ Cb2,
                                             const _Float16* __restrict__ Cw3p,
                                             const float* __restrict__ Cb3,
                                             float* __restrict__ out) {
  __shared__ __align__(32) _Float16 X1[128 * Hn];  // 64 KB
  __shared__ __align__(32) _Float16 X2[128 * Hn];  // 64 KB
  const int r0 = blockIdx.x * 128;
  const int m0 = (threadIdx.x >> 5) * 16;
  const int l  = threadIdx.x & 31;
  const int nn = l & 15, mo = (l & 16) ? 8 : 0;

  // ---- actor path ----
  layer256(latent, r0 + m0, Hn, Aw1, Ab1, X1, m0);
  __syncthreads();
  layer256(X1, m0, Hn, Aw2, Ab2, X2, m0);
  __syncthreads();
  {
    v8f acc = {};
#pragma unroll
    for (int k0 = 0; k0 < Hn; k0 += 32)
      acc = wmma16(frag_a_f16(X2, m0, k0, Hn), frag_b_w(Aw3p, 0, k0, Hn), acc);
    if (nn < An) {
      float bv = Ab3[nn];
#pragma unroll
      for (int e = 0; e < 8; ++e)
        out[(size_t)(r0 + m0 + e + mo) * (An + 1) + nn] = acc[e] + bv;
    }
  }
  __syncthreads();

  // ---- critic path (reuse LDS buffers) ----
  layer256(latent, r0 + m0, Hn, Cw1, Cb1, X1, m0);
  __syncthreads();
  layer256(X1, m0, Hn, Cw2, Cb2, X2, m0);
  __syncthreads();
  {
    v8f acc = {};
#pragma unroll
    for (int k0 = 0; k0 < Hn; k0 += 32)
      acc = wmma16(frag_a_f16(X2, m0, k0, Hn), frag_b_w(Cw3p, 0, k0, Hn), acc);
    if (nn == 0) {
      float bv = Cb3[0];
#pragma unroll
      for (int e = 0; e < 8; ++e)
        out[(size_t)(r0 + m0 + e + mo) * (An + 1) + An] = acc[e] + bv;
    }
  }
}

// ---------------- launch -----------------------------------------------------

extern "C" void kernel_launch(void* const* d_in, const int* in_sizes, int n_in,
                              void* d_out, int out_size, void* d_ws, size_t ws_size,
                              hipStream_t stream) {
  (void)in_sizes; (void)n_in; (void)out_size; (void)ws_size;
  const float* feat    = (const float*)d_in[0];
  const float* hidden  = (const float*)d_in[1];
  const float* W_ih    = (const float*)d_in[2];
  const float* W_hh    = (const float*)d_in[3];
  const float* b_ih    = (const float*)d_in[4];
  const float* b_hh    = (const float*)d_in[5];
  const float* Aw1     = (const float*)d_in[6];
  const float* Ab1     = (const float*)d_in[7];
  const float* Aw2     = (const float*)d_in[8];
  const float* Ab2     = (const float*)d_in[9];
  const float* Aw3     = (const float*)d_in[10];
  const float* Ab3     = (const float*)d_in[11];
  const float* Cw1     = (const float*)d_in[12];
  const float* Cb1     = (const float*)d_in[13];
  const float* Cw2     = (const float*)d_in[14];
  const float* Cb2     = (const float*)d_in[15];
  const float* Cw3     = (const float*)d_in[16];
  const float* Cb3     = (const float*)d_in[17];
  const float* log_std = (const float*)d_in[18];
  const int*   starts  = (const int*)d_in[19];

  float* outp = (float*)d_out;
  const size_t OUT_MAIN = (size_t)Bn * Tn * (An + 1);
  float* out_std = outp + OUT_MAIN;
  float* out_h   = outp + OUT_MAIN + An;

  // workspace carve-up (256B aligned)
  char* ws = (char*)d_ws;
  size_t off = 0;
  auto alloc16 = [&](size_t elems) {
    _Float16* p = (_Float16*)(ws + off);
    off += ((elems * sizeof(_Float16) + 255) / 256) * 256;
    return p;
  };
  _Float16* Wih16 = alloc16((size_t)3 * Hn * Fn);
  _Float16* Whh16 = alloc16((size_t)3 * Hn * Hn);
  _Float16* Aw1h  = alloc16((size_t)Hn * Hn);
  _Float16* Aw2h  = alloc16((size_t)Hn * Hn);
  _Float16* Aw3h  = alloc16((size_t)16 * Hn);   // padded 12->16 rows
  _Float16* Cw1h  = alloc16((size_t)Hn * Hn);
  _Float16* Cw2h  = alloc16((size_t)Hn * Hn);
  _Float16* Cw3h  = alloc16((size_t)16 * Hn);   // padded 1->16 rows
  _Float16* xg    = alloc16((size_t)Bn * Tn * 3 * Hn);   // ~403 MB
  _Float16* lat   = alloc16((size_t)Bn * Tn * Hn);       // ~134 MB

  auto conv = [&](const float* s, _Float16* d, int rows_src, int rows_dst, int cols) {
    int total = rows_dst * cols;
    k_convert_pad<<<(total + 255) / 256, 256, 0, stream>>>(s, d, rows_src, cols, total);
  };
  conv(W_ih, Wih16, 3 * Hn, 3 * Hn, Fn);
  conv(W_hh, Whh16, 3 * Hn, 3 * Hn, Hn);
  conv(Aw1, Aw1h, Hn, Hn, Hn);
  conv(Aw2, Aw2h, Hn, Hn, Hn);
  conv(Aw3, Aw3h, An, 16, Hn);
  conv(Cw1, Cw1h, Hn, Hn, Hn);
  conv(Cw2, Cw2h, Hn, Hn, Hn);
  conv(Cw3, Cw3h, 1, 16, Hn);

  k_std<<<1, 32, 0, stream>>>(log_std, out_std);

  dim3 gxg((Bn * Tn) / 128, (3 * Hn) / 128);
  k_xg<<<gxg, 256, 0, stream>>>(feat, Wih16, b_ih, xg);

  k_scan<<<Bn / 32, 256, 0, stream>>>(feat ? xg : xg, Whh16, b_hh, hidden, starts, lat, out_h);

  k_mlp<<<(Bn * Tn) / 128, 256, 0, stream>>>(lat, Aw1h, Ab1, Aw2h, Ab2, Aw3h, Ab3,
                                             Cw1h, Cb1, Cw2h, Cb2, Cw3h, Cb3, outp);
}
